// C3SNN_ModelT_7224134992419
// MI455X (gfx1250) — matrix operand order repaired and checked by reading, not compiled, not run on hardware
//
#include <hip/hip_runtime.h>
#include <hip/hip_bf16.h>

typedef __attribute__((ext_vector_type(16))) _Float16 v16h;
typedef __attribute__((ext_vector_type(8)))  _Float16 v8h;
typedef __attribute__((ext_vector_type(8)))  float    v8f;
typedef __attribute__((ext_vector_type(4)))  unsigned int u32x4;
typedef __attribute__((ext_vector_type(8)))  int      i32x8;
typedef __attribute__((ext_vector_type(4)))  int      i32x4;

#define BATCH 1024
#define SEQ 32

#ifndef __has_builtin
#define __has_builtin(x) 0
#endif
#if __has_builtin(__builtin_amdgcn_tensor_load_to_lds) && \
    __has_builtin(__builtin_amdgcn_s_wait_tensorcnt)
#define USE_TDM 1
#else
#define USE_TDM 0
#endif

#define WMMA_F16(A, B, C) \
    __builtin_amdgcn_wmma_f32_16x16x32_f16(false, (A), false, (B), (short)0, (C), false, false)

// LDS layout for the staged weight tile: TDM pads 4 dwords (8 halves) after
// every 256 dwords (512 halves) stored -> breaks the 16-way bank conflict of a
// K-stride row layout while keeping every fragment 16-byte aligned.
__device__ __forceinline__ int lds_woff(int linear_halves) {
    return linear_halves + ((linear_halves >> 9) << 3);
}

__device__ __forceinline__ v16h cat16(v8h lo, v8h hi) {
    return __builtin_shufflevector(lo, hi,
        0, 1, 2, 3, 4, 5, 6, 7, 8, 9, 10, 11, 12, 13, 14, 15);
}

// ---------------- f32 -> f16 weight conversion ----------------
__global__ void f32_to_f16_kernel(const float* __restrict__ src,
                                  _Float16* __restrict__ dst, int n) {
    int i = blockIdx.x * blockDim.x + threadIdx.x;
    if (i < n) dst[i] = (_Float16)src[i];
}

// conv weights [Cout,Cin,3,3] f32 -> [tap][oc][ci] f16 (B-fragment friendly)
__global__ void wconv_perm_kernel(const float* __restrict__ src,
                                  _Float16* __restrict__ dst,
                                  int Cout, int Cin) {
    int idx = blockIdx.x * blockDim.x + threadIdx.x;
    int total = 9 * Cout * Cin;
    if (idx >= total) return;
    int ci = idx % Cin; int t = idx / Cin;
    int oc = t % Cout;  int tap = t / Cout;
    dst[idx] = (_Float16)src[(oc * Cin + ci) * 9 + tap];
}

// ---------------- zero init ----------------
__global__ void zero_kernel(float* __restrict__ p, int n) {
    int i = blockIdx.x * blockDim.x + threadIdx.x;
    if (i < n) p[i] = 0.0f;
}

// ---------------- conv1 (Cin=3, K too small for WMMA): scalar, emits NHWC f16 ----
__global__ void conv1_nhwc_kernel(const float* __restrict__ in,
                                  const float* __restrict__ w,
                                  const float* __restrict__ bias,
                                  _Float16* __restrict__ out) {
    int idx = blockIdx.x * blockDim.x + threadIdx.x;   // [B,16,16,32] NHWC order
    if (idx >= BATCH * 16 * 16 * 32) return;
    int oc = idx & 31; int t = idx >> 5;
    int ox = t & 15;   t >>= 4;
    int oy = t & 15;   int b = t >> 4;

    float m = 0.0f;
    const float* wb = w + oc * 27;
    #pragma unroll
    for (int py = 0; py < 2; ++py) {
        #pragma unroll
        for (int px = 0; px < 2; ++px) {
            int y = 2 * oy + py, x = 2 * ox + px;
            float acc = bias[oc];
            #pragma unroll
            for (int ci = 0; ci < 3; ++ci) {
                const float* irow = in + ((b * 3 + ci) * 32) * 32;
                #pragma unroll
                for (int ky = 0; ky < 3; ++ky) {
                    int iy = y + ky - 1;
                    if ((unsigned)iy >= 32u) continue;
                    #pragma unroll
                    for (int kx = 0; kx < 3; ++kx) {
                        int ix = x + kx - 1;
                        if ((unsigned)ix >= 32u) continue;
                        acc = fmaf(irow[iy * 32 + ix], wb[ci * 9 + ky * 3 + kx], acc);
                    }
                }
            }
            m = fmaxf(m, fmaxf(acc, 0.0f));
        }
    }
    out[idx] = (_Float16)m;
}

// ---------------- conv2 via implicit GEMM (9 shifted WMMA GEMMs, K=32/tap) ----
// in:  [B,16,16,32] NHWC f16 ; wt: [9][64][32] f16 ; out: [B,8,8,64] NHWC f16
// One wave per (b, pooled row): M = 2 conv rows x 16 cols, N = 64.
__global__ void __launch_bounds__(256)
conv2_wmma_kernel(const _Float16* __restrict__ xin,
                  const _Float16* __restrict__ wt,
                  const float* __restrict__ bias,
                  _Float16* __restrict__ xout) {
    const int wave = threadIdx.x >> 5;
    const int lane = threadIdx.x & 31;
    const int lrow = lane & 15;
    const int hi16 = lane >> 4;
    int task = blockIdx.x * 8 + wave;      // BATCH*8 tasks
    int oy = task & 7, b = task >> 3;
    const int y0 = oy * 2;

    v8f acc[2][4] = {};
    #pragma unroll
    for (int dy = 0; dy < 3; ++dy) {
        #pragma unroll
        for (int dx = 0; dx < 3; ++dx) {
            const int tap = dy * 3 + dx;
            const _Float16* bb = wt + tap * 2048 + lrow * 32 + hi16 * 16;
            v16h bf[4];
            #pragma unroll
            for (int nt = 0; nt < 4; ++nt) bf[nt] = *(const v16h*)(bb + nt * 512);

            const int xs = lrow + dx - 1;
            const bool xok = (unsigned)xs < 16u;
            #pragma unroll
            for (int r2 = 0; r2 < 2; ++r2) {
                int ys = y0 + r2 + dy - 1;
                v16h a = {};
                if (((unsigned)ys < 16u) && xok) {
                    const _Float16* ap =
                        xin + ((size_t)(b * 16 + ys) * 16 + xs) * 32 + hi16 * 16;
                    a = cat16(*(const v8h*)ap, *(const v8h*)(ap + 8));
                }
                #pragma unroll
                for (int nt = 0; nt < 4; ++nt)
                    acc[r2][nt] = WMMA_F16(a, bf[nt], acc[r2][nt]);
            }
        }
    }
    // bias + relu + 2x2 maxpool (x-pairs live in adjacent C rows of this lane)
    #pragma unroll
    for (int nt = 0; nt < 4; ++nt) {
        int oc = nt * 16 + lrow;
        float bn = bias[oc];
        #pragma unroll
        for (int q = 0; q < 4; ++q) {
            float p = fmaxf(fmaxf(fmaxf(acc[0][nt][2 * q] + bn, 0.0f),
                                  fmaxf(acc[0][nt][2 * q + 1] + bn, 0.0f)),
                            fmaxf(fmaxf(acc[1][nt][2 * q] + bn, 0.0f),
                                  fmaxf(acc[1][nt][2 * q + 1] + bn, 0.0f)));
            int ox = q + 4 * hi16;
            xout[((size_t)(b * 8 + oy) * 8 + ox) * 64 + oc] = (_Float16)p;
        }
    }
}

// ---------------- conv3 via implicit GEMM (K=64/tap) -> feat = 4*pool ----
// in: [B,8,8,64] NHWC f16 ; wt: [9][64][64] f16 ; feat: [B,1024] fp32 NCHW-flat
// One wave per (b, pooled row): M-tile packs 2 conv rows x 8 cols (hi16 = y).
__global__ void __launch_bounds__(256)
conv3_wmma_kernel(const _Float16* __restrict__ xin,
                  const _Float16* __restrict__ wt,
                  const float* __restrict__ bias,
                  float* __restrict__ feat) {
    const int wave = threadIdx.x >> 5;
    const int lane = threadIdx.x & 31;
    const int lrow = lane & 15;
    const int hi16 = lane >> 4;
    int task = blockIdx.x * 8 + wave;      // BATCH*4 tasks
    int oy = task & 3, b = task >> 2;
    const int y0 = oy * 2;
    const int yl = lrow >> 3, xr = lrow & 7;   // A row m=lrow -> (y_local, x)

    v8f acc[4] = {};
    #pragma unroll
    for (int dy = 0; dy < 3; ++dy) {
        #pragma unroll
        for (int dx = 0; dx < 3; ++dx) {
            const int tap = dy * 3 + dx;
            const int ys = y0 + yl + dy - 1;
            const int xs = xr + dx - 1;
            const bool ok = ((unsigned)ys < 8u) && ((unsigned)xs < 8u);
            const _Float16* ap =
                xin + ((size_t)(b * 8 + (ok ? ys : 0)) * 8 + (ok ? xs : 0)) * 64;
            const _Float16* bb = wt + tap * 4096 + lrow * 64 + hi16 * 16;
            #pragma unroll
            for (int kc = 0; kc < 64; kc += 32) {
                v16h a = {};
                if (ok) {
                    const _Float16* app = ap + kc + hi16 * 16;
                    a = cat16(*(const v8h*)app, *(const v8h*)(app + 8));
                }
                #pragma unroll
                for (int nt = 0; nt < 4; ++nt) {
                    v16h bf = *(const v16h*)(bb + nt * 1024 + kc);
                    acc[nt] = WMMA_F16(a, bf, acc[nt]);
                }
            }
        }
    }
    // bias + relu + 2x2 pool: x-pairs per lane, y-pairs across lane^16
    #pragma unroll
    for (int nt = 0; nt < 4; ++nt) {
        int oc = nt * 16 + lrow;
        float bn = bias[oc];
        #pragma unroll
        for (int q = 0; q < 4; ++q) {
            float mx = fmaxf(fmaxf(acc[nt][2 * q] + bn, 0.0f),
                             fmaxf(acc[nt][2 * q + 1] + bn, 0.0f));
            float other = __shfl_xor(mx, 16, 32);
            float p = fmaxf(mx, other);
            if (hi16 == 0)
                feat[(size_t)b * 1024 + oc * 16 + oy * 4 + q] = 4.0f * p;
        }
    }
}

// ---------------- ConstantCurrentLIF encoder step ----------------
__global__ void encoder_kernel(const float* __restrict__ feat,
                               float* __restrict__ ve,
                               _Float16* __restrict__ z, int n) {
    int i = blockIdx.x * blockDim.x + threadIdx.x;
    if (i >= n) return;
    float v = 0.9f * ve[i] + 0.1f * feat[i];
    float s = (v - 1.0f) > 0.0f ? 1.0f : 0.0f;
    ve[i] = v * (1.0f - s);
    z[i] = (_Float16)s;
}

// ---------------- WMMA GEMM (cur = Z @ W^T + b) fused with LIF cell ----------------
__global__ void __launch_bounds__(256)
gemm_lif_kernel(const _Float16* __restrict__ Zh,
                const _Float16* __restrict__ Wh,
                const float* __restrict__ bias,
                float* __restrict__ vstate,
                float* __restrict__ istate,
                _Float16* __restrict__ sout,
                int K, int N) {
    __shared__ __attribute__((aligned(64))) _Float16 wtile[16896];

    const int wave = threadIdx.x >> 5;
    const int lane = threadIdx.x & 31;
    const int lrow = lane & 15;
    const int hi16 = lane >> 4;

    const int n0 = blockIdx.x * 16;
    const int m_base = (blockIdx.y * 8 + wave) * 64;

    // ---- stage weight tile W[n0 .. n0+15][0..K) into LDS via TDM ----
#if USE_TDM
    if (threadIdx.x < 32) {
        unsigned long long ga =
            (unsigned long long)(uintptr_t)(Wh + (size_t)n0 * K);
        unsigned lds_base = (unsigned)(uintptr_t)(&wtile[0]);
        u32x4 g0;
        g0[0] = 1u;                                   // count=1 valid descriptor
        g0[1] = lds_base;                             // lds_addr
        g0[2] = (unsigned)(ga & 0xffffffffull);       // global_addr[31:0]
        g0[3] = (unsigned)((ga >> 32) & 0x1ffffffull) // global_addr[56:32]
              | (2u << 30);                           // type=2 (image)
        i32x8 g1;
        g1[0] = (int)((1u << 16)      // data_size = 2 bytes
                    | (1u << 20)      // pad_enable
                    | (7u << 22)      // pad_interval = 256 dwords
                    | (3u << 25));    // pad_amount  = 4 dwords
        g1[1] = (int)((K & 0xffff) << 16);
        g1[2] = (int)(((unsigned)K >> 16) | ((N & 0xffff) << 16));
        g1[3] = (int)(((unsigned)N >> 16) | ((K & 0xffff) << 16));
        g1[4] = 16;                    // tile_dim1 = 16 rows
        g1[5] = K;                     // tensor_dim0_stride = K
        g1[6] = 0;
        g1[7] = 0;
        i32x4 z4 = {0, 0, 0, 0};
#if __clang_major__ <= 22
        __builtin_amdgcn_tensor_load_to_lds(g0, g1, z4, z4, 0);
#else
        i32x8 z8 = {0, 0, 0, 0, 0, 0, 0, 0};
        __builtin_amdgcn_tensor_load_to_lds(g0, g1, z4, z4, z8, 0);
#endif
        __builtin_amdgcn_s_wait_tensorcnt(0);
    }
    __syncthreads();
#else
    for (int idx = threadIdx.x * 8; idx < 16 * K; idx += 256 * 8) {
        int r = idx / K, c = idx % K;
        v8h v = *(const v8h*)(Wh + (size_t)(n0 + r) * K + c);
        *(v8h*)(wtile + lds_woff(idx)) = v;
    }
    __syncthreads();
#endif

    v8f acc0 = {}, acc1 = {}, acc2 = {}, acc3 = {};

    const _Float16* zrow = Zh + (size_t)(m_base + lrow) * K + hi16 * 16;
    const size_t zstride = (size_t)16 * K;
    const int blin = lrow * K + hi16 * 16;

    for (int k0 = 0; k0 < K; k0 += 32) {
        const _Float16* bp = wtile + lds_woff(blin + k0);
        v16h bfrag = cat16(*(const v8h*)bp, *(const v8h*)(bp + 8));

        __builtin_prefetch(zrow + k0 + 128, 0, 3);   // WGP-scope stream-ahead

        const _Float16* zp = zrow + k0;
        #pragma unroll
        for (int t = 0; t < 4; ++t) {
            v16h afrag = cat16(*(const v8h*)zp, *(const v8h*)(zp + 8));
            if (t == 0)      acc0 = WMMA_F16(afrag, bfrag, acc0);
            else if (t == 1) acc1 = WMMA_F16(afrag, bfrag, acc1);
            else if (t == 2) acc2 = WMMA_F16(afrag, bfrag, acc2);
            else             acc3 = WMMA_F16(afrag, bfrag, acc3);
            zp += zstride;
        }
    }

    const int n = n0 + lrow;
    const float bn = bias[n];
    #pragma unroll
    for (int t = 0; t < 4; ++t) {
        v8f acc = (t == 0) ? acc0 : (t == 1) ? acc1 : (t == 2) ? acc2 : acc3;
        int m0 = m_base + 16 * t + hi16 * 8;
        #pragma unroll
        for (int r = 0; r < 8; ++r) {
            size_t idx = (size_t)(m0 + r) * N + n;
            float cur = acc[r] + bn;
            float vo = vstate[idx];
            float io = istate[idx];
            float vd = 0.9f * vo + 0.1f * io;
            float s  = (vd - 0.4f) > 0.0f ? 1.0f : 0.0f;
            vstate[idx] = (1.0f - s) * vd;
            istate[idx] = 0.8f * io + cur;
            sout[idx]   = (_Float16)s;
        }
    }
}

// ---------------- LILinear decoder step ----------------
__global__ void li_kernel(const _Float16* __restrict__ s2,
                          const float* __restrict__ liw,
                          float* __restrict__ vl,
                          float* __restrict__ il,
                          float* __restrict__ out) {
    int idx = blockIdx.x * blockDim.x + threadIdx.x;
    if (idx >= BATCH * 10) return;
    int b = idx / 10, c = idx % 10;
    const _Float16* sp = s2 + (size_t)b * 256;
    const float* wp = liw + (size_t)c * 256;
    float acc = 0.0f;
    #pragma unroll 8
    for (int k = 0; k < 256; ++k) acc = fmaf((float)sp[k], wp[k], acc);
    float ij = il[idx] + acc;
    float v  = 0.9f * vl[idx] + 0.1f * ij;
    vl[idx] = v;
    il[idx] = 0.8f * ij;
    out[idx] = v;
}

// ---------------- launch ----------------
extern "C" void kernel_launch(void* const* d_in, const int* in_sizes, int n_in,
                              void* d_out, int out_size, void* d_ws, size_t ws_size,
                              hipStream_t stream) {
    const float* x     = (const float*)d_in[0];
    const float* w1    = (const float*)d_in[1];
    const float* b1    = (const float*)d_in[2];
    const float* w2    = (const float*)d_in[3];
    const float* b2    = (const float*)d_in[4];
    const float* w3    = (const float*)d_in[5];
    const float* b3    = (const float*)d_in[6];
    const float* fc1w  = (const float*)d_in[7];
    const float* fc1b  = (const float*)d_in[8];
    const float* fc2w  = (const float*)d_in[9];
    const float* fc2b  = (const float*)d_in[10];
    const float* liw   = (const float*)d_in[11];

    char* ws = (char*)d_ws;
    // persistent region
    float*    feat  = (float*)   (ws + 0);                 // 4,194,304 B
    _Float16* fc1wh = (_Float16*)(ws + 4194304);           // 1,048,576 B
    _Float16* fc2wh = (_Float16*)(ws + 5242880);           //   262,144 B
    const size_t UNION = 5505024;
    // conv scratch (used only before the SNN loop)
    _Float16* x2h = (_Float16*)(ws + UNION);               // 16,777,216 B
    _Float16* x3h = (_Float16*)(ws + UNION + 16777216);    //  8,388,608 B
    _Float16* w2t = (_Float16*)(ws + UNION + 25165824);    //     36,864 B
    _Float16* w3t = (_Float16*)(ws + UNION + 25202688);    //     73,728 B
    // SNN state (overlaps conv scratch; initialized after conv)
    float* state = (float*)(ws + UNION);
    float* ve = state;                                     // 1024*1024
    float* v1 = ve + 1048576;                              // 1024*512
    float* i1 = v1 + 524288;
    float* v2 = i1 + 524288;                               // 1024*256
    float* i2 = v2 + 262144;
    float* vl = i2 + 262144;                               // 1024*10
    float* il = vl + 10240;
    const int N_STATE = 1048576 + 2 * 524288 + 2 * 262144 + 2 * 10240;
    _Float16* z_h  = (_Float16*)(ws + 16072704);           // right after SNN state
    _Float16* s1_h = (_Float16*)(ws + 18169856);
    _Float16* s2_h = (_Float16*)(ws + 19218432);

    // NOTE: z/s buffers at 16-19 MB overlap x2h tail; both are transient but
    // used in disjoint phases only if conv finishes first -- it does: all conv
    // kernels are enqueued before any SNN kernel on the same stream.

    // weights -> f16
    f32_to_f16_kernel<<<(524288 + 255) / 256, 256, 0, stream>>>(fc1w, fc1wh, 524288);
    f32_to_f16_kernel<<<(131072 + 255) / 256, 256, 0, stream>>>(fc2w, fc2wh, 131072);
    wconv_perm_kernel<<<(18432 + 255) / 256, 256, 0, stream>>>(w2, w2t, 64, 32);
    wconv_perm_kernel<<<(36864 + 255) / 256, 256, 0, stream>>>(w3, w3t, 64, 64);

    // CNN feature extractor
    conv1_nhwc_kernel<<<(BATCH * 16 * 16 * 32 + 255) / 256, 256, 0, stream>>>(
        x, w1, b1, x2h);
    conv2_wmma_kernel<<<BATCH * 8 / 8, 256, 0, stream>>>(x2h, w2t, b2, x3h);
    conv3_wmma_kernel<<<BATCH * 4 / 8, 256, 0, stream>>>(x3h, w3t, b3, feat);

    // zero all recurrent state
    zero_kernel<<<(N_STATE + 255) / 256, 256, 0, stream>>>(state, N_STATE);

    // 32-timestep SNN loop
    for (int t = 0; t < SEQ; ++t) {
        encoder_kernel<<<(BATCH * 1024 + 255) / 256, 256, 0, stream>>>(
            feat, ve, z_h, BATCH * 1024);
        gemm_lif_kernel<<<dim3(512 / 16, 2), 256, 0, stream>>>(
            z_h, fc1wh, fc1b, v1, i1, s1_h, 1024, 512);
        gemm_lif_kernel<<<dim3(256 / 16, 2), 256, 0, stream>>>(
            s1_h, fc2wh, fc2b, v2, i2, s2_h, 512, 256);
        li_kernel<<<(BATCH * 10 + 255) / 256, 256, 0, stream>>>(
            s2_h, liw, vl, il, (float*)d_out);
    }
}
